// CachedTransformerEncoderLayer_74706661147038
// MI455X (gfx1250) — matrix operand order repaired and checked by reading, not compiled, not run on hardware
//
#include <hip/hip_runtime.h>

// ---------------------------------------------------------------------------
// CDNA5 / gfx1250 transformer encoder layer with cached-KV recompute.
// All GEMMs via v_wmma_f32_16x16x32_bf16 (bf16 in, f32 accumulate).
// Attention V-tiles staged via GLOBAL_LOAD_ASYNC_TO_LDS (ASYNCcnt) when the
// toolchain exposes the gfx1250 async builtins.
// ---------------------------------------------------------------------------

typedef __attribute__((ext_vector_type(16))) __bf16 v16bf;
typedef __attribute__((ext_vector_type(8)))  float  v8f;
typedef int v4i __attribute__((vector_size(16)));   // matches async builtin param

#define AS1 __attribute__((address_space(1)))
#define AS3 __attribute__((address_space(3)))

#define WMMA_BF16(A, B, C) \
  __builtin_amdgcn_wmma_f32_16x16x32_bf16(false, (A), false, (B), (short)0, (C), false, false)

union FragBF {
  v16bf v;
  __bf16 e[16];
  uint4 u[2];
};

__device__ __forceinline__ __bf16 f2bf(float f) {
  union { float f; unsigned u; } x; x.f = f;
  unsigned r = x.u + 0x7FFFu + ((x.u >> 16) & 1u);   // round-to-nearest-even
  unsigned short h = (unsigned short)(r >> 16);
  __bf16 b;
  __builtin_memcpy(&b, &h, 2);
  return b;
}

__device__ __forceinline__ v8f vzero8() {
  v8f z = {0.f, 0.f, 0.f, 0.f, 0.f, 0.f, 0.f, 0.f};
  return z;
}

// ---------------------------------------------------------------------------
// f32 -> bf16 elementwise convert
// ---------------------------------------------------------------------------
__global__ void cvt_f32_bf16(const float* __restrict__ in, __bf16* __restrict__ out, long n) {
  long i = (long)blockIdx.x * blockDim.x + threadIdx.x;
  long stride = (long)gridDim.x * blockDim.x;
  for (; i < n; i += stride) out[i] = f2bf(in[i]);
}

// ---------------------------------------------------------------------------
// Generic GEMM:  Y[M,N] = X[M,K] @ W[N,K]^T + bias, optional relu.
//   mode 0: store f32 row-major to outF
//   mode 1: store bf16 row-major to outB
//   mode 2: scatter bf16 into [B,H,S,hd] cache via row_idx (KV recompute)
//   gatherA: A row rI maps to X row (rI/R)*S_tot + row_idx[rI%R]
// Wave-level: each wave owns a 32x64 output strip: two A-row groups share the
// same 4 B fragments -> 12 b128 loads per 8 WMMAs per K=32 step.
// Block = 128 threads (4 waves) -> 128x64 tile.
// ---------------------------------------------------------------------------
__global__ void gemm_bf16_wmma(const __bf16* __restrict__ X,
                               const __bf16* __restrict__ W,
                               const float* __restrict__ bias,
                               int M, int N, int K,
                               int mode, int relu,
                               __bf16* __restrict__ outB,
                               float* __restrict__ outF,
                               const int* __restrict__ row_idx,
                               int gatherA, int R, int S_tot,
                               int H, int hd) {
  const int lane = threadIdx.x & 31;
  const int wave = threadIdx.x >> 5;
  const int half = lane >> 4;       // 0: lanes 0-15, 1: lanes 16-31
  const int l16  = lane & 15;
  const int row0 = blockIdx.y * 128 + wave * 32;
  const int col0 = blockIdx.x * 64;

  // Per-lane A-matrix rows (16-bit A layout: M = lane&15 for both halves).
  int   rI[2];
  bool  rv[2];
  long  xrow[2];
  rI[0] = row0 + l16;
  rI[1] = row0 + 16 + l16;
#pragma unroll
  for (int g = 0; g < 2; ++g) {
    rv[g] = (rI[g] < M);
    xrow[g] = rI[g];
    if (gatherA) {
      int bb = rI[g] / R;
      int rr = rI[g] - bb * R;
      if (rv[g]) xrow[g] = (long)bb * S_tot + row_idx[rr];
    }
  }

  v8f acc[2][4];
#pragma unroll
  for (int g = 0; g < 2; ++g)
#pragma unroll
    for (int nt = 0; nt < 4; ++nt) acc[g][nt] = vzero8();

  const __bf16* aBase0 = X + xrow[0] * (long)K + half * 8;
  const __bf16* aBase1 = X + xrow[1] * (long)K + half * 8;

  for (int k0 = 0; k0 < K; k0 += 32) {
    FragBF a0, a1;
    if (rv[0]) {
      const uint4* p = (const uint4*)(aBase0 + k0);
      a0.u[0] = p[0];           // K = k0 + half*8 + 0..7
      a0.u[1] = p[2];           // K = k0 + half*8 + 16..23
      __builtin_prefetch(aBase0 + k0 + 64, 0, 1);
    } else {
      uint4 z; z.x = z.y = z.z = z.w = 0u;
      a0.u[0] = z; a0.u[1] = z;
    }
    if (rv[1]) {
      const uint4* p = (const uint4*)(aBase1 + k0);
      a1.u[0] = p[0];
      a1.u[1] = p[2];
    } else {
      uint4 z; z.x = z.y = z.z = z.w = 0u;
      a1.u[0] = z; a1.u[1] = z;
    }
#pragma unroll
    for (int nt = 0; nt < 4; ++nt) {
      const int n = col0 + nt * 16 + l16;           // B layout: column = lane&15
      FragBF b;
      const uint4* q = (const uint4*)(W + (long)n * K + k0 + half * 8);
      b.u[0] = q[0];            // K = k0 + half*8 + 0..7
      b.u[1] = q[2];            // K = k0 + half*8 + 16..23
      acc[0][nt] = WMMA_BF16(a0.v, b.v, acc[0][nt]);
      acc[1][nt] = WMMA_BF16(a1.v, b.v, acc[1][nt]);
    }
  }

  // Epilogue. C layout: lane half h, VGPR j -> row m = j + h*8, col = lane&15.
#pragma unroll
  for (int g = 0; g < 2; ++g) {
#pragma unroll
    for (int nt = 0; nt < 4; ++nt) {
      const int n = col0 + nt * 16 + l16;
      const float bn = bias ? bias[n] : 0.f;
#pragma unroll
      for (int j = 0; j < 8; ++j) {
        const int orow = row0 + g * 16 + j + half * 8;
        if (orow >= M) continue;
        float y = acc[g][nt][j] + bn;
        if (relu) y = fmaxf(y, 0.f);
        if (mode == 0) {
          outF[(long)orow * N + n] = y;
        } else if (mode == 1) {
          outB[(long)orow * N + n] = f2bf(y);
        } else {
          // scatter into bf16 KV cache [B,H,S_tot,hd]
          int bb = orow / R;
          int rr = orow - bb * R;
          int sI = row_idx[rr];
          int hh = n / hd;
          int dd = n - hh * hd;
          outB[(((long)bb * H + hh) * S_tot + sI) * hd + dd] = f2bf(y);
        }
      }
    }
  }
}

// ---------------------------------------------------------------------------
// Attention: one wave per (b,h, 16-row q block). Full 16xS score strip lives
// in LDS (S=2048 -> 128KB, fits the CDNA5 320KB WGP LDS). Exact softmax with
// wave32 shfl reductions, then P@V via WMMA with V staged through LDS using
// gfx1250 async global->LDS loads when available.
// q layout [B*S, D] (col = h*hd+d); k/v layout [B,H,S,hd]; out [B*S, D].
// ---------------------------------------------------------------------------
__global__ void attn_wmma(const __bf16* __restrict__ qm,
                          const __bf16* __restrict__ km,
                          const __bf16* __restrict__ vm,
                          __bf16* __restrict__ outm,
                          int H, int S, int D, int hd) {
  extern __shared__ char smem[];
  float* sc = (float*)smem;                                   // [16][S]
  __bf16* vt = (__bf16*)(smem + (size_t)16 * S * sizeof(float)); // [32][72]
  const int VPAD = 72;

  const int lane = threadIdx.x & 31;
  const int half = lane >> 4;
  const int l16  = lane & 15;
  const int qb = blockIdx.x;            // q tile (16 rows)
  const int bh = blockIdx.y;            // b*H + h
  const int b = bh / H;
  const int h = bh - b * H;
  const float scale = rsqrtf((float)hd);

  // --- Load Q fragments (16 x 64 -> two A frags, K=0..31 and 32..63) ---
  FragBF aq[2];
  {
    const __bf16* qp = qm + ((long)(b * S + qb * 16 + l16)) * D + h * hd + half * 8;
#pragma unroll
    for (int t = 0; t < 2; ++t) {
      const uint4* p = (const uint4*)(qp + t * 32);
      aq[t].u[0] = p[0];
      aq[t].u[1] = p[2];
    }
  }

  // --- Phase 1: scores = Q @ K^T * scale into LDS ---
  const __bf16* kbase = km + (long)bh * S * hd;
  for (int kt = 0; kt < (S >> 4); ++kt) {
    const __bf16* kp = kbase + (long)(kt * 16 + l16) * hd + half * 8;
    FragBF bk0, bk1;
    { const uint4* p = (const uint4*)kp;        bk0.u[0] = p[0]; bk0.u[1] = p[2]; }
    { const uint4* p = (const uint4*)(kp + 32); bk1.u[0] = p[0]; bk1.u[1] = p[2]; }
    v8f s = vzero8();
    s = WMMA_BF16(aq[0].v, bk0.v, s);
    s = WMMA_BF16(aq[1].v, bk1.v, s);
#pragma unroll
    for (int j = 0; j < 8; ++j)
      sc[(j + half * 8) * S + kt * 16 + l16] = s[j] * scale;
  }
  __syncthreads();

  // --- Phase 2: exact softmax per row (wave32 reductions) ---
  for (int r = 0; r < 16; ++r) {
    float mx = -3.402823e38f;
    for (int c = lane; c < S; c += 32) mx = fmaxf(mx, sc[r * S + c]);
#pragma unroll
    for (int o = 16; o > 0; o >>= 1) mx = fmaxf(mx, __shfl_xor(mx, o, 32));
    float sum = 0.f;
    for (int c = lane; c < S; c += 32) {
      float e = __expf(sc[r * S + c] - mx);
      sc[r * S + c] = e;
      sum += e;
    }
#pragma unroll
    for (int o = 16; o > 0; o >>= 1) sum += __shfl_xor(sum, o, 32);
    const float inv = 1.f / sum;
    for (int c = lane; c < S; c += 32) sc[r * S + c] *= inv;
  }
  __syncthreads();

  // --- Phase 3: attn = P @ V via WMMA (stage V tiles through LDS) ---
  v8f acc[4];
  acc[0] = vzero8(); acc[1] = vzero8(); acc[2] = vzero8(); acc[3] = vzero8();
  const __bf16* vbase = vm + (long)bh * S * hd;

  for (int kb = 0; kb < S; kb += 32) {
    {
      const __bf16* gsrc = vbase + (long)(kb + lane) * hd;   // row kb+lane
      __bf16* ldst = vt + lane * VPAD;
#if __has_builtin(__builtin_amdgcn_global_load_async_to_lds_b128)
      // gfx1250 async global->LDS DMA, tracked by ASYNCcnt (no VGPR staging).
#pragma unroll
      for (int t = 0; t < 4; ++t)
        __builtin_amdgcn_global_load_async_to_lds_b128(
            (AS1 v4i*)(gsrc + t * 8),
            (AS3 v4i*)(ldst + t * 8),
            0, 0);
#if __has_builtin(__builtin_amdgcn_s_wait_asynccnt)
      __builtin_amdgcn_s_wait_asynccnt(0);
#endif
#else
      const uint4* srcq = (const uint4*)gsrc;
      uint4* dstq = (uint4*)ldst;
#pragma unroll
      for (int t = 0; t < 4; ++t) dstq[t] = srcq[t];
#endif
    }
    __syncthreads();

    // A frag from probs: row = lane&15, K = kb + {half*8+0..7, 16+half*8+0..7}
    FragBF pa;
    {
      const float* pr = sc + l16 * S + kb + half * 8;
#pragma unroll
      for (int e = 0; e < 8; ++e) pa.e[e] = f2bf(pr[e]);
#pragma unroll
      for (int e = 0; e < 8; ++e) pa.e[8 + e] = f2bf(pr[16 + e]);
    }
#pragma unroll
    for (int nt = 0; nt < 4; ++nt) {
      FragBF vb;   // B frag: col = nt*16 + lane&15, K = {half*8+0..7, 16+half*8+0..7}
#pragma unroll
      for (int e = 0; e < 16; ++e)
        vb.e[e] = vt[(half * 8 + (e & 7) + (e >> 3) * 16) * VPAD + nt * 16 + l16];
      acc[nt] = WMMA_BF16(pa.v, vb.v, acc[nt]);
    }
    __syncthreads();
  }

  // --- Store attn (transposed back to [B*S, D], col = h*hd + d) ---
#pragma unroll
  for (int nt = 0; nt < 4; ++nt)
#pragma unroll
    for (int j = 0; j < 8; ++j)
      outm[((long)(b * S + qb * 16 + j + half * 8)) * D + h * hd + nt * 16 + l16] =
          f2bf(acc[nt][j]);
}

// ---------------------------------------------------------------------------
// LayerNorm(a + b) * g + beta  -> f32 (and optional bf16 copy)
// ---------------------------------------------------------------------------
__global__ void ln_residual(const float* __restrict__ a,
                            const float* __restrict__ bres,
                            const float* __restrict__ g,
                            const float* __restrict__ be,
                            float* __restrict__ outF,
                            __bf16* __restrict__ outB,
                            int D) {
  __shared__ float red[256];
  const int row = blockIdx.x;
  const float* pa = a + (long)row * D;
  const float* pb = bres + (long)row * D;

  float s = 0.f;
  for (int i = threadIdx.x; i < D; i += 256) s += pa[i] + pb[i];
  red[threadIdx.x] = s;
  __syncthreads();
  for (int o = 128; o > 0; o >>= 1) {
    if (threadIdx.x < o) red[threadIdx.x] += red[threadIdx.x + o];
    __syncthreads();
  }
  const float mu = red[0] / D;
  __syncthreads();

  float v = 0.f;
  for (int i = threadIdx.x; i < D; i += 256) {
    float x = pa[i] + pb[i] - mu;
    v += x * x;
  }
  red[threadIdx.x] = v;
  __syncthreads();
  for (int o = 128; o > 0; o >>= 1) {
    if (threadIdx.x < o) red[threadIdx.x] += red[threadIdx.x + o];
    __syncthreads();
  }
  const float inv = rsqrtf(red[0] / D + 1e-5f);

  for (int i = threadIdx.x; i < D; i += 256) {
    float y = (pa[i] + pb[i] - mu) * inv * g[i] + be[i];
    outF[(long)row * D + i] = y;
    if (outB) outB[(long)row * D + i] = f2bf(y);
  }
}

// ---------------------------------------------------------------------------
// Host side
// ---------------------------------------------------------------------------
extern "C" void kernel_launch(void* const* d_in, const int* in_sizes, int n_in,
                              void* d_out, int out_size, void* d_ws, size_t ws_size,
                              hipStream_t stream) {
  (void)in_sizes; (void)n_in; (void)out_size; (void)ws_size;
  constexpr int Bn = 2, S = 2048, D = 1024, H = 16, HD = 64, FF = 4096, R = 409;
  constexpr int BS = Bn * S;            // 4096
  constexpr long KV = (long)Bn * H * S * HD; // == BS*D

  const float* src = (const float*)d_in[0];
  const int*   ridx = (const int*)d_in[1];
  const float* ck  = (const float*)d_in[2];
  const float* cv  = (const float*)d_in[3];
  const float* Wq = (const float*)d_in[4];  const float* bq = (const float*)d_in[5];
  const float* Wk = (const float*)d_in[6];  const float* bk = (const float*)d_in[7];
  const float* Wv = (const float*)d_in[8];  const float* bv = (const float*)d_in[9];
  const float* Wo = (const float*)d_in[10]; const float* bo = (const float*)d_in[11];
  const float* W1 = (const float*)d_in[12]; const float* b1 = (const float*)d_in[13];
  const float* W2 = (const float*)d_in[14]; const float* b2 = (const float*)d_in[15];
  const float* g1 = (const float*)d_in[16]; const float* be1 = (const float*)d_in[17];
  const float* g2 = (const float*)d_in[18]; const float* be2 = (const float*)d_in[19];
  float* out = (float*)d_out;

  // Bump allocator over workspace (256B aligned).
  char* wp = (char*)d_ws;
  auto alloc = [&](size_t bytes) -> char* {
    char* p = wp;
    wp += (bytes + 255) & ~(size_t)255;
    return p;
  };
  __bf16* srcB  = (__bf16*)alloc((size_t)BS * D * 2);
  __bf16* WqB   = (__bf16*)alloc((size_t)D * D * 2);
  __bf16* WkB   = (__bf16*)alloc((size_t)D * D * 2);
  __bf16* WvB   = (__bf16*)alloc((size_t)D * D * 2);
  __bf16* WoB   = (__bf16*)alloc((size_t)D * D * 2);
  __bf16* W1B   = (__bf16*)alloc((size_t)FF * D * 2);
  __bf16* W2B   = (__bf16*)alloc((size_t)D * FF * 2);
  __bf16* qB    = (__bf16*)alloc((size_t)BS * D * 2);
  __bf16* kB    = (__bf16*)alloc((size_t)KV * 2);
  __bf16* vB    = (__bf16*)alloc((size_t)KV * 2);
  __bf16* attnB = (__bf16*)alloc((size_t)BS * D * 2);
  float*  attnO = (float*)alloc((size_t)BS * D * 4);
  float*  x1    = (float*)alloc((size_t)BS * D * 4);
  __bf16* x1B   = (__bf16*)alloc((size_t)BS * D * 2);
  __bf16* ffB   = (__bf16*)alloc((size_t)BS * FF * 2);
  float*  ff2   = (float*)alloc((size_t)BS * D * 4);

  // 1) Convert activations / weights / KV cache to bf16.
  auto cvt = [&](const float* in, __bf16* o, long n) {
    int blocks = (int)((n + 255 * 8) / (256 * 8));
    if (blocks > 2048) blocks = 2048;
    if (blocks < 1) blocks = 1;
    cvt_f32_bf16<<<blocks, 256, 0, stream>>>(in, o, n);
  };
  cvt(src, srcB, (long)BS * D);
  cvt(Wq, WqB, (long)D * D);
  cvt(Wk, WkB, (long)D * D);
  cvt(Wv, WvB, (long)D * D);
  cvt(Wo, WoB, (long)D * D);
  cvt(W1, W1B, (long)FF * D);
  cvt(W2, W2B, (long)D * FF);
  cvt(ck, kB, KV);
  cvt(cv, vB, KV);

  auto gemm = [&](const __bf16* X, const __bf16* W, const float* bias,
                  int M, int N, int K, int mode, int relu,
                  __bf16* oB, float* oF, const int* idx, int gatherA) {
    dim3 grid((N + 63) / 64, (M + 127) / 128);
    gemm_bf16_wmma<<<grid, 128, 0, stream>>>(X, W, bias, M, N, K, mode, relu,
                                             oB, oF, idx, gatherA, R, S, H, HD);
  };

  // 2) Q projection: [BS,D] bf16.
  gemm(srcB, WqB, bq, BS, D, D, /*mode=*/1, 0, qB, nullptr, nullptr, 0);

  // 3) K/V recompute: gather src rows by idx, project, scatter into bf16 cache.
  gemm(srcB, WkB, bk, Bn * R, D, D, /*mode=*/2, 0, kB, nullptr, ridx, 1);
  gemm(srcB, WvB, bv, Bn * R, D, D, /*mode=*/2, 0, vB, nullptr, ridx, 1);

  // 4) Attention (full-row softmax in 128KB LDS strip + WMMA P@V).
  {
    size_t shmem = (size_t)16 * S * sizeof(float) + (size_t)32 * 72 * 2;
    dim3 grid(S / 16, Bn * H);
    attn_wmma<<<grid, 32, shmem, stream>>>(qB, kB, vB, attnB, H, S, D, HD);
  }

  // 5) Output projection -> f32.
  gemm(attnB, WoB, bo, BS, D, D, /*mode=*/0, 0, nullptr, attnO, nullptr, 0);

  // 6) LN1(src + attnO) -> x1 (f32) + x1 (bf16).
  ln_residual<<<BS, 256, 0, stream>>>(src, attnO, g1, be1, x1, x1B, D);

  // 7) FFN: relu(x1 @ W1^T + b1) -> bf16, then @ W2^T + b2 -> f32.
  gemm(x1B, W1B, b1, BS, FF, D, /*mode=*/1, /*relu=*/1, ffB, nullptr, nullptr, 0);
  gemm(ffB, W2B, b2, BS, D, FF, /*mode=*/0, 0, nullptr, ff2, nullptr, 0);

  // 8) LN2(x1 + ff2) -> output (f32).
  ln_residual<<<BS, 256, 0, stream>>>(x1, ff2, g2, be2, out, nullptr, D);
}
